// Mamba_Qformer_36206574305380
// MI455X (gfx1250) — matrix operand order repaired and checked by reading, not compiled, not run on hardware
//
#include <hip/hip_runtime.h>
#include <hip/hip_bf16.h>
#include <math.h>

typedef __bf16 bf16t;
typedef __attribute__((ext_vector_type(16))) bf16t v16bf;
typedef __attribute__((ext_vector_type(8)))  float v8f;
typedef __attribute__((ext_vector_type(4)))  float v4f;

#define NLAYERS 32
#define BB   2
#define L0   256
#define LQ   7
#define LC   263      // 256 + 7 query tokens
#define DIN  1408
#define DM   512
#define DI   1024
#define DTR  32
#define DS   16
#define M0   (BB*L0)  // 512
#define MC   (BB*LC)  // 526

static __device__ __forceinline__ float siluf(float x)     { return x / (1.0f + __expf(-x)); }
static __device__ __forceinline__ float softplusf(float x) { return (x > 20.0f) ? x : log1pf(__expf(x)); }

// ---------------------------------------------------------------------------
// Generic GEMM: C[M x N] = A[M x K (lda)] * W[N x K]^T  (+bias, +softplus)
// bf16 WMMA 16x16x32, fp32 accumulate. One wave computes a 16x64 strip
// (4 accumulators). Per k-step: load+convert A and all 4 B fragments first,
// then issue the 4 WMMAs back-to-back (avoids WMMA->VALU WAR hazard NOPs,
// enables co-execution of the WMMA chain with next iteration's loads).
// grid = (N/64, ceil(M/16)), block = 32 (wave32).
// mode: 0 = plain, 1 = +bias, 2 = softplus(x + bias)
// ---------------------------------------------------------------------------
__global__ void gemm_wmma_bf16(const float* __restrict__ A, const float* __restrict__ W,
                               const float* __restrict__ bias, float* __restrict__ C,
                               int M, int K, int lda, int ldc, int mode)
{
  const int lane = threadIdx.x;        // 0..31
  const int m    = lane & 15;
  const int half = lane >> 4;
  const int row0 = blockIdx.y << 4;
  const int col0 = blockIdx.x << 6;    // 64 output columns per wave

  // Clamp: rows >= M compute garbage that is never stored (store is guarded).
  int arow = row0 + m; if (arow >= M) arow = M - 1;
  const float* __restrict__ Arow = A + (size_t)arow * lda;
  const float* __restrict__ Wr0  = W + (size_t)(col0 + m) * K;

  v8f acc[4] = {};

  for (int k0 = 0; k0 < K; k0 += 32) {
    // ---- Phase 1: load + convert A fragment and all four B fragments ----
    // A fragment, contiguous k order (this code's choice): lanes 0-15 hold K 0..15, lanes 16-31 hold K 16..31
    const v4f* Ap0 = (const v4f*)(Arow + k0 + (half << 4));
    const v4f* Ap1 = (const v4f*)(Arow + k0 + 8 + (half << 4));
    v4f a0 = Ap0[0], a1 = Ap0[1], a2 = Ap1[0], a3 = Ap1[1];
    v16bf a;
    #pragma unroll
    for (int j = 0; j < 4; ++j) {
      a[j]      = (bf16t)a0[j];
      a[4 + j]  = (bf16t)a1[j];
      a[8 + j]  = (bf16t)a2[j];
      a[12 + j] = (bf16t)a3[j];
    }

    v16bf bfr[4];
    #pragma unroll
    for (int t = 0; t < 4; ++t) {
      // B fragment, same contiguous k order as A: lanes 0-15 hold K 0..15, lanes 16-31 hold K 16..31; lane&15 = N
      const v4f* Wp = (const v4f*)(Wr0 + (size_t)(t * 16) * K + k0 + (half << 4));
      v4f w0 = Wp[0], w1 = Wp[1], w2 = Wp[2], w3 = Wp[3];
      #pragma unroll
      for (int j = 0; j < 4; ++j) {
        bfr[t][j]      = (bf16t)w0[j];
        bfr[t][4 + j]  = (bf16t)w1[j];
        bfr[t][8 + j]  = (bf16t)w2[j];
        bfr[t][12 + j] = (bf16t)w3[j];
      }
    }

    // ---- Phase 2: four back-to-back WMMAs on independent accumulators ----
    #pragma unroll
    for (int t = 0; t < 4; ++t)
      acc[t] = __builtin_amdgcn_wmma_f32_16x16x32_bf16(false, a, false, bfr[t], (short)0,
                                                       acc[t], false, false);
  }

  // C/D layout: VGPR v, lanes 0-15 -> M = v, lanes 16-31 -> M = 8+v; N = lane&15
  #pragma unroll
  for (int t = 0; t < 4; ++t) {
    int c = col0 + t * 16 + m;
    #pragma unroll
    for (int v = 0; v < 8; ++v) {
      int r = row0 + (half << 3) + v;
      if (r < M) {
        float val = acc[t][v];
        if (mode >= 1) val += bias[c];
        if (mode == 2) val = softplusf(val);
        C[(size_t)r * ldc + c] = val;
      }
    }
  }
}

// ---------------------------------------------------------------------------
// Build hs = concat(proj_out, query) over the token axis; zero residual.
// ---------------------------------------------------------------------------
__global__ void assemble_kernel(const float* __restrict__ proj, const float* __restrict__ query,
                                float* __restrict__ hs, float* __restrict__ res)
{
  int idx = blockIdx.x * blockDim.x + threadIdx.x;
  if (idx >= MC * DM) return;
  int d = idx % DM;
  int r = idx / DM;
  int l = r % LC, b = r / LC;
  float v = (l < L0) ? proj[(size_t)(b * L0 + l) * DM + d]
                     : query[(size_t)(l - L0) * DM + d];
  hs[idx]  = v;
  res[idx] = 0.0f;
}

// ---------------------------------------------------------------------------
// res += hs;  xln = LayerNorm(res) * lw + lb    (one block per row, 512 cols)
// ---------------------------------------------------------------------------
__global__ void add_ln_kernel(const float* __restrict__ hs, float* __restrict__ res,
                              const float* __restrict__ lw, const float* __restrict__ lb,
                              float* __restrict__ xln)
{
  __shared__ float s1[256], s2[256];
  int r = blockIdx.x, t = threadIdx.x;
  size_t base = (size_t)r * DM;
  float v0 = hs[base + t]       + res[base + t];
  float v1 = hs[base + t + 256] + res[base + t + 256];
  res[base + t] = v0; res[base + t + 256] = v1;
  s1[t] = v0 + v1; s2[t] = v0 * v0 + v1 * v1;
  __syncthreads();
  for (int s = 128; s > 0; s >>= 1) {
    if (t < s) { s1[t] += s1[t + s]; s2[t] += s2[t + s]; }
    __syncthreads();
  }
  float mu  = s1[0] * (1.0f / DM);
  float var = s2[0] * (1.0f / DM) - mu * mu;
  float inv = rsqrtf(var + 1e-5f);
  xln[base + t]       = (v0 - mu) * inv * lw[t]       + lb[t];
  xln[base + t + 256] = (v1 - mu) * inv * lw[t + 256] + lb[t + 256];
}

// ---------------------------------------------------------------------------
// Depth-wise causal conv (kernel 4) over tokens + SiLU. xs = xz[..., :DI].
// ---------------------------------------------------------------------------
__global__ void conv_silu_kernel(const float* __restrict__ xz, const float* __restrict__ cw,
                                 const float* __restrict__ cb, float* __restrict__ xc)
{
  int idx = blockIdx.x * blockDim.x + threadIdx.x;
  if (idx >= MC * DI) return;
  int c = idx % DI;
  int r = idx / DI;
  int l = r % LC, b = r / LC;
  float acc = cb[c];
  #pragma unroll
  for (int k = 0; k < 4; ++k) {
    int ll = l + k - 3;
    if (ll >= 0) acc += cw[c * 4 + k] * xz[(size_t)(b * LC + ll) * (2 * DI) + c];
  }
  xc[idx] = siluf(acc);
}

// ---------------------------------------------------------------------------
// Selective scan: one thread per (batch, channel); B_t/C_t staged in LDS.
// y = (scan + D*xc) * silu(z)
// ---------------------------------------------------------------------------
__global__ void scan_kernel(const float* __restrict__ dbc, const float* __restrict__ dt,
                            const float* __restrict__ xc, const float* __restrict__ xz,
                            const float* __restrict__ A_log, const float* __restrict__ D_skip,
                            float* __restrict__ y)
{
  __shared__ float sB[DS], sC[DS];
  int d = blockIdx.x * blockDim.x + threadIdx.x;   // channel
  int b = blockIdx.y;
  float a[DS], h[DS];
  #pragma unroll
  for (int n = 0; n < DS; ++n) { a[n] = -__expf(A_log[(size_t)d * DS + n]); h[n] = 0.0f; }
  float dsk = D_skip[d];
  for (int l = 0; l < LC; ++l) {
    int r = b * LC + l;
    if (threadIdx.x < 2 * DS) {
      int n = threadIdx.x;
      float v = dbc[(size_t)r * (DTR + 2 * DS) + DTR + n];
      if (n < DS) sB[n] = v; else sC[n - DS] = v;
    }
    __syncthreads();
    float dtv = dt[(size_t)r * DI + d];
    float xv  = xc[(size_t)r * DI + d];
    float acc = 0.0f;
    #pragma unroll
    for (int n = 0; n < DS; ++n) {
      h[n] = __expf(dtv * a[n]) * h[n] + dtv * sB[n] * xv;
      acc += h[n] * sC[n];
    }
    acc += dsk * xv;
    float z = xz[(size_t)r * (2 * DI) + DI + d];
    acc *= siluf(z);
    y[(size_t)r * DI + d] = acc;
    __syncthreads();
  }
}

// ---------------------------------------------------------------------------
// Final: out = LN(hs + res)[:, -LQ:]  (one block per output row)
// ---------------------------------------------------------------------------
__global__ void final_ln_kernel(const float* __restrict__ hs, const float* __restrict__ res,
                                const float* __restrict__ nw, const float* __restrict__ nb,
                                float* __restrict__ out)
{
  __shared__ float s1[256], s2[256];
  int q = blockIdx.x % LQ, b = blockIdx.x / LQ;
  int r = b * LC + (LC - LQ) + q;
  int t = threadIdx.x;
  size_t base = (size_t)r * DM;
  float v0 = hs[base + t]       + res[base + t];
  float v1 = hs[base + t + 256] + res[base + t + 256];
  s1[t] = v0 + v1; s2[t] = v0 * v0 + v1 * v1;
  __syncthreads();
  for (int s = 128; s > 0; s >>= 1) {
    if (t < s) { s1[t] += s1[t + s]; s2[t] += s2[t + s]; }
    __syncthreads();
  }
  float mu  = s1[0] * (1.0f / DM);
  float var = s2[0] * (1.0f / DM) - mu * mu;
  float inv = rsqrtf(var + 1e-5f);
  size_t o = (size_t)(b * LQ + q) * DM;
  out[o + t]       = (v0 - mu) * inv * nw[t]       + nb[t];
  out[o + t + 256] = (v1 - mu) * inv * nw[t + 256] + nb[t + 256];
}

// ---------------------------------------------------------------------------
extern "C" void kernel_launch(void* const* d_in, const int* in_sizes, int n_in,
                              void* d_out, int out_size, void* d_ws, size_t ws_size,
                              hipStream_t stream)
{
  (void)in_sizes; (void)n_in; (void)out_size; (void)ws_size;
  const float* hidden  = (const float*)d_in[0];
  const float* proj_w  = (const float*)d_in[1];
  const float* proj_b  = (const float*)d_in[2];
  const float* query   = (const float*)d_in[3];
  const float* ln_w    = (const float*)d_in[4];
  const float* ln_b    = (const float*)d_in[5];
  const float* in_w    = (const float*)d_in[6];
  const float* conv_w  = (const float*)d_in[7];
  const float* conv_b  = (const float*)d_in[8];
  const float* xproj_w = (const float*)d_in[9];
  const float* dt_w    = (const float*)d_in[10];
  const float* dt_b    = (const float*)d_in[11];
  const float* A_log   = (const float*)d_in[12];
  const float* D_skip  = (const float*)d_in[13];
  const float* out_w   = (const float*)d_in[14];
  const float* norm_w  = (const float*)d_in[15];
  const float* norm_b  = (const float*)d_in[16];
  float* out = (float*)d_out;

  float* ws   = (float*)d_ws;
  float* hs   = ws; ws += MC * DM;
  float* res  = ws; ws += MC * DM;
  float* xln  = ws; ws += MC * DM;
  float* tmp  = ws; ws += M0 * DM;
  float* xz   = ws; ws += (size_t)MC * 2 * DI;
  float* xc   = ws; ws += (size_t)MC * DI;
  float* dbc  = ws; ws += (size_t)MC * (DTR + 2 * DS);
  float* dtb  = ws; ws += (size_t)MC * DI;
  float* ybuf = ws; ws += (size_t)MC * DI;

  const dim3 blk32(32);
  const int mt = (MC + 15) / 16;   // 33 row tiles

  // initial projection: (B*L0 x DIN) x (DM x DIN)^T + bias
  gemm_wmma_bf16<<<dim3(DM / 64, M0 / 16), blk32, 0, stream>>>(
      hidden, proj_w, proj_b, tmp, M0, DIN, DIN, DM, 1);
  assemble_kernel<<<(MC * DM + 255) / 256, 256, 0, stream>>>(tmp, query, hs, res);

  for (int layer = 0; layer < NLAYERS; ++layer) {
    const float* lw  = ln_w    + (size_t)layer * DM;
    const float* lb  = ln_b    + (size_t)layer * DM;
    const float* iw  = in_w    + (size_t)layer * 2 * DI * DM;
    const float* cw  = conv_w  + (size_t)layer * DI * 4;
    const float* cb  = conv_b  + (size_t)layer * DI;
    const float* xw  = xproj_w + (size_t)layer * (DTR + 2 * DS) * DI;
    const float* dw  = dt_w    + (size_t)layer * DI * DTR;
    const float* db  = dt_b    + (size_t)layer * DI;
    const float* al  = A_log   + (size_t)layer * DI * DS;
    const float* dsk = D_skip  + (size_t)layer * DI;
    const float* ow  = out_w   + (size_t)layer * DM * DI;

    add_ln_kernel<<<MC, 256, 0, stream>>>(hs, res, lw, lb, xln);
    gemm_wmma_bf16<<<dim3(2 * DI / 64, mt), blk32, 0, stream>>>(
        xln, iw, nullptr, xz, MC, DM, DM, 2 * DI, 0);
    conv_silu_kernel<<<(MC * DI + 255) / 256, 256, 0, stream>>>(xz, cw, cb, xc);
    gemm_wmma_bf16<<<dim3((DTR + 2 * DS) / 64, mt), blk32, 0, stream>>>(
        xc, xw, nullptr, dbc, MC, DI, DI, DTR + 2 * DS, 0);
    gemm_wmma_bf16<<<dim3(DI / 64, mt), blk32, 0, stream>>>(
        dbc, dw, db, dtb, MC, DTR, DTR + 2 * DS, DI, 2);     // softplus(x + dt_b)
    scan_kernel<<<dim3(DI / 256, BB), 256, 0, stream>>>(dbc, dtb, xc, xz, al, dsk, ybuf);
    gemm_wmma_bf16<<<dim3(DM / 64, mt), blk32, 0, stream>>>(
        ybuf, ow, nullptr, hs, MC, DI, DI, DM, 0);
  }

  final_ln_kernel<<<BB * LQ, 256, 0, stream>>>(hs, res, norm_w, norm_b, out);
}
